// LlaMoeBlock_48318382080220
// MI455X (gfx1250) — matrix operand order repaired and checked by reading, not compiled, not run on hardware
//
#include <hip/hip_runtime.h>

// ---------------- WMMA helpers (gfx1250, wave32) ----------------
typedef __attribute__((ext_vector_type(16))) __bf16 v16bf;
typedef __attribute__((ext_vector_type(8)))  float  v8f;
typedef __attribute__((ext_vector_type(4)))  unsigned int u32x4;

union CFrag { v8f v; float f[8]; };

// f32 -> bf16 conversions: use V_CVT_PK_BF16_F32 (1 VALU op per 2 converts)
#define USE_CVT_ASM 1
#if USE_CVT_ASM
__device__ __forceinline__ unsigned int pack2bf(float x, float y) {
  unsigned int r;
  asm("v_cvt_pk_bf16_f32 %0, %1, %2" : "=v"(r) : "v"(x), "v"(y));
  return r;
}
__device__ __forceinline__ unsigned short f2bf(float x) {
  unsigned int r;
  asm("v_cvt_pk_bf16_f32 %0, %1, 0" : "=v"(r) : "v"(x));
  return (unsigned short)r;
}
#else
__device__ __forceinline__ unsigned short f2bf(float x) {
  unsigned int u = __float_as_uint(x);
  unsigned int r = ((u >> 16) & 1u) + 0x7FFFu;   // round-to-nearest-even
  return (unsigned short)((u + r) >> 16);
}
__device__ __forceinline__ unsigned int pack2bf(float x, float y) {
  return (unsigned int)f2bf(x) | ((unsigned int)f2bf(y) << 16);
}
#endif
__device__ __forceinline__ float bf2f(unsigned short b) {
  return __uint_as_float(((unsigned int)b) << 16);
}
__device__ __forceinline__ v8f wmma_bf16(v16bf a, v16bf b, v8f c) {
  return __builtin_amdgcn_wmma_f32_16x16x32_bf16(false, a, false, b, (short)0, c, false, false);
}

// ---- CDNA5 async global->LDS copy (ASYNCcnt-tracked, no VGPR round trip) ----
#define USE_ASYNC_LDS 1
__device__ __forceinline__ void stage_b128(unsigned short* lds_dst,
                                           const unsigned short* gsrc) {
#if USE_ASYNC_LDS
  asm volatile("global_load_async_to_lds_b128 %0, %1, off"
               :: "v"((unsigned int)(unsigned long long)(uintptr_t)lds_dst),
                  "v"((unsigned long long)(uintptr_t)gsrc)
               : "memory");
#else
  *(u32x4*)lds_dst = *(const u32x4*)gsrc;
#endif
}
__device__ __forceinline__ void stage_wait() {
#if USE_ASYNC_LDS
  asm volatile("s_wait_asynccnt 0" ::: "memory");
#endif
}

// A-operand fragment: per lane K = half*16 + i, contiguous. 2x b128 loads.
__device__ __forceinline__ v16bf load_frag_a(const unsigned short* rowbase, int half) {
  union { v16bf v; u32x4 q[2]; } f;
  const u32x4* p = (const u32x4*)(rowbase + half * 16);
  f.q[0] = p[0];
  f.q[1] = p[1];
  return f.v;
}
// B-operand fragment: per lane K = half*16 + i, contiguous. 2x b128 loads.
__device__ __forceinline__ v16bf load_frag_b(const unsigned short* colbase, int half) {
  union { v16bf v; u32x4 q[2]; } f;
  const u32x4* p = (const u32x4*)(colbase + half * 16);
  f.q[0] = p[0];
  f.q[1] = p[1];
  return f.v;
}

// ---------------- problem constants ----------------
#define S_LEN 2048
#define D_DIM 2048
#define F_DIM 4096
#define NH    16
#define DHD   128
#define NEXP  8

// ---------------- GEMM tiling ----------------
#define TM 128
#define TN 64
#define TK 32
#define LDA_S 40   // 80B row stride (16B aligned)
#define LDB_T 40   // 80B col stride (16B aligned)

// ============================================================
// rmsnorm (f32 in) -> bf16 out
// ============================================================
__global__ __launch_bounds__(256) void rmsnorm_cast_kernel(
    const float* __restrict__ x, const float* __restrict__ w,
    unsigned short* __restrict__ out)
{
  int row = blockIdx.x;
  size_t base = (size_t)row * D_DIM;
  float ss = 0.f;
#pragma unroll
  for (int i = 0; i < D_DIM / 256; ++i) {
    float v = x[base + threadIdx.x + i * 256];
    ss += v * v;
  }
  __shared__ float red[8];
#pragma unroll
  for (int off = 16; off > 0; off >>= 1) ss += __shfl_down(ss, off, 32);
  if ((threadIdx.x & 31) == 0) red[threadIdx.x >> 5] = ss;
  __syncthreads();
  if (threadIdx.x == 0) {
    float tot = 0.f;
    for (int i = 0; i < 8; ++i) tot += red[i];
    red[0] = rsqrtf(tot / (float)D_DIM + 1e-5f);
  }
  __syncthreads();
  float rinv = red[0];
#pragma unroll
  for (int i = 0; i < D_DIM / 256; ++i) {
    int c = threadIdx.x + i * 256;
    out[base + c] = f2bf(x[base + c] * rinv * w[c]);
  }
}

// ============================================================
// residual add + rmsnorm: h2 = res + o (f32), xb2 = rmsnorm(h2)*w (bf16)
// ============================================================
__global__ __launch_bounds__(256) void add_rmsnorm_kernel(
    const float* __restrict__ res, const float* __restrict__ o,
    const float* __restrict__ w, float* __restrict__ h2,
    unsigned short* __restrict__ xb2)
{
  int row = blockIdx.x;
  size_t base = (size_t)row * D_DIM;
  float ss = 0.f;
#pragma unroll
  for (int i = 0; i < D_DIM / 256; ++i) {
    int c = threadIdx.x + i * 256;
    float v = res[base + c] + o[base + c];
    h2[base + c] = v;
    ss += v * v;
  }
  __shared__ float red[8];
#pragma unroll
  for (int off = 16; off > 0; off >>= 1) ss += __shfl_down(ss, off, 32);
  if ((threadIdx.x & 31) == 0) red[threadIdx.x >> 5] = ss;
  __syncthreads();
  if (threadIdx.x == 0) {
    float tot = 0.f;
    for (int i = 0; i < 8; ++i) tot += red[i];
    red[0] = rsqrtf(tot / (float)D_DIM + 1e-5f);
  }
  __syncthreads();
  float rinv = red[0];
#pragma unroll
  for (int i = 0; i < D_DIM / 256; ++i) {
    int c = threadIdx.x + i * 256;
    xb2[base + c] = f2bf(h2[base + c] * rinv * w[c]);
  }
}

// ============================================================
// Generic WMMA GEMM: C[M,N] f32 = A[M,K] bf16 * B[K,N] f32(->bf16)
// ============================================================
__global__ __launch_bounds__(256) void gemm_bf16_kernel(
    const unsigned short* __restrict__ A, const float* __restrict__ B,
    float* __restrict__ C, int N, int K)
{
  __shared__ unsigned short As[TM * LDA_S];
  __shared__ unsigned short Bt[TN * LDB_T];
  const int tid = threadIdx.x;
  const int wave = tid >> 5, lane = tid & 31;
  const int half = lane >> 4, nn = lane & 15;
  const int m0 = blockIdx.x * TM, n0 = blockIdx.y * TN;

  const int ar = tid >> 2, aq = tid & 3;
  const unsigned short* asrc0 = A + (size_t)(m0 + ar) * K + aq * 8;
  const unsigned short* asrc1 = A + (size_t)(m0 + ar + 64) * K + aq * 8;
  unsigned short* adst0 = As + ar * LDA_S + aq * 8;
  unsigned short* adst1 = As + (ar + 64) * LDA_S + aq * 8;

  const int nB = tid & 63, r2b = tid >> 6;
  const float* bsrc = B + (size_t)(2 * r2b) * N + n0 + nB;
  unsigned int* bdst_base = (unsigned int*)(Bt + nB * LDB_T + 2 * r2b);

  CFrag acc[4];
#pragma unroll
  for (int t = 0; t < 4; ++t)
#pragma unroll
    for (int j = 0; j < 8; ++j) acc[t].f[j] = 0.f;

  for (int k0 = 0; k0 < K; k0 += TK) {
    stage_b128(adst0, asrc0);
    stage_b128(adst1, asrc1);
    asrc0 += TK; asrc1 += TK;
    unsigned int* bdst = bdst_base;
#pragma unroll
    for (int it = 0; it < 4; ++it) {
      float x = bsrc[0], y = bsrc[N];
      *bdst = pack2bf(x, y);
      bsrc += (size_t)8 * N;
      bdst += 4;
    }
    if (k0 + TK < K) __builtin_prefetch(bsrc, 0, 1);
    stage_wait();
    __syncthreads();
    v16bf a = load_frag_a(As + (wave * 16 + nn) * LDA_S, half);
#pragma unroll
    for (int t = 0; t < 4; ++t) {
      v16bf b = load_frag_b(Bt + (t * 16 + nn) * LDB_T, half);
      acc[t].v = wmma_bf16(a, b, acc[t].v);
    }
    __syncthreads();
  }
  float* crow = C + (size_t)(m0 + wave * 16 + 8 * half) * N + n0 + nn;
#pragma unroll
  for (int r = 0; r < 8; ++r) {
#pragma unroll
    for (int t = 0; t < 4; ++t) crow[t * 16] = acc[t].f[r];
    crow += N;
  }
}

// ============================================================
// RoPE + split qkv[S,3D] f32 -> qb/kb/vb [h][s][dh] bf16
// ============================================================
__global__ __launch_bounds__(64) void rope_split_kernel(
    const float* __restrict__ qkv, unsigned short* __restrict__ qb,
    unsigned short* __restrict__ kb, unsigned short* __restrict__ vb)
{
  const int halfd = DHD / 2;
  int s = blockIdx.x, h = blockIdx.y, d = threadIdx.x;
  const float* base = qkv + (size_t)s * 3 * D_DIM;
  float inv = __powf(10000.f, -(float)(2 * d) / (float)DHD);
  float ang = (float)s * inv;
  float sn = __sinf(ang), cs = __cosf(ang);
  size_t ob = ((size_t)h * S_LEN + s) * DHD;
  float q0 = base[h * DHD + d], q1 = base[h * DHD + d + halfd];
  qb[ob + d]         = f2bf(q0 * cs - q1 * sn);
  qb[ob + d + halfd] = f2bf(q1 * cs + q0 * sn);
  float k0 = base[D_DIM + h * DHD + d], k1 = base[D_DIM + h * DHD + d + halfd];
  kb[ob + d]         = f2bf(k0 * cs - k1 * sn);
  kb[ob + d + halfd] = f2bf(k1 * cs + k0 * sn);
  vb[ob + d]         = f2bf(base[2 * D_DIM + h * DHD + d]);
  vb[ob + d + halfd] = f2bf(base[2 * D_DIM + h * DHD + d + halfd]);
}

// ============================================================
// Causal flash attention, bf16 WMMA for QK^T and PV.
// ============================================================
#define VSW_S 72   // 144B stride, 16B aligned
__global__ __launch_bounds__(128) void flash_attn_kernel(
    const unsigned short* __restrict__ qb, const unsigned short* __restrict__ kb,
    const unsigned short* __restrict__ vb, unsigned short* __restrict__ ctxb)
{
  const float scale = 0.08838834764831845f;  // 1/sqrt(128)
  int h = blockIdx.y;
  int q0 = blockIdx.x * 64;
  int wave = threadIdx.x >> 5, lane = threadIdx.x & 31;
  int half = lane >> 4, nn = lane & 15;
  int qrow0 = q0 + wave * 16;
  const unsigned short* Q  = qb + (size_t)h * S_LEN * DHD;
  const unsigned short* Kp = kb + (size_t)h * S_LEN * DHD;
  const unsigned short* Vp = vb + (size_t)h * S_LEN * DHD;

  v16bf qf[4];
#pragma unroll
  for (int kk = 0; kk < 4; ++kk)
    qf[kk] = load_frag_a(Q + (size_t)(qrow0 + nn) * DHD + kk * 32, half);

  CFrag O[8];
  float mrow[8], lrow[8];
#pragma unroll
  for (int t = 0; t < 8; ++t)
#pragma unroll
    for (int j = 0; j < 8; ++j) O[t].f[j] = 0.f;
#pragma unroll
  for (int r = 0; r < 8; ++r) { mrow[r] = -3.0e38f; lrow[r] = 0.f; }

  __shared__ unsigned short Vsw[DHD * VSW_S];      // [d][64 keys]
  __shared__ unsigned short Pst[4][16 * VSW_S];    // per wave [m][64 keys]

  const int vq = threadIdx.x & 15, vk0 = threadIdx.x >> 4;

  int jend = q0 + 64;  // causal bound, uniform per block
  for (int j0 = 0; j0 < jend; j0 += 64) {
    __syncthreads();
    {
      const u32x4* vsrc = (const u32x4*)(Vp + (size_t)(j0 + vk0) * DHD) + vq;
      unsigned short* vdst = Vsw + (vq * 8) * VSW_S + vk0;
#pragma unroll
      for (int it = 0; it < 8; ++it) {
        u32x4 vv = *vsrc;
        const unsigned short* pu = (const unsigned short*)&vv;
#pragma unroll
        for (int jj = 0; jj < 8; ++jj) vdst[jj * VSW_S] = pu[jj];
        vsrc += 8 * DHD / 8;
        vdst += 8;
      }
    }
    __syncthreads();

    CFrag sacc[4];
#pragma unroll
    for (int t = 0; t < 4; ++t)
#pragma unroll
      for (int j = 0; j < 8; ++j) sacc[t].f[j] = 0.f;
#pragma unroll
    for (int kk = 0; kk < 4; ++kk) {
#pragma unroll
      for (int t = 0; t < 4; ++t) {
        int key = j0 + t * 16 + nn;
        v16bf b = load_frag_b(Kp + (size_t)key * DHD + kk * 32, half);
        sacc[t].v = wmma_bf16(qf[kk], b, sacc[t].v);
      }
    }
    float rowmax[8];
#pragma unroll
    for (int r = 0; r < 8; ++r) rowmax[r] = -3.0e38f;
#pragma unroll
    for (int t = 0; t < 4; ++t)
#pragma unroll
      for (int r = 0; r < 8; ++r) {
        int key = j0 + t * 16 + nn;
        int qidx = qrow0 + r + 8 * half;
        float sc = sacc[t].f[r] * scale;
        if (key > qidx) sc = -1.0e30f;
        sacc[t].f[r] = sc;
        rowmax[r] = fmaxf(rowmax[r], sc);
      }
#pragma unroll
    for (int r = 0; r < 8; ++r)
#pragma unroll
      for (int m = 1; m < 16; m <<= 1)
        rowmax[r] = fmaxf(rowmax[r], __shfl_xor(rowmax[r], m, 32));

    float ps[8];
#pragma unroll
    for (int r = 0; r < 8; ++r) {
      float mnew = fmaxf(mrow[r], rowmax[r]);
      float corr = __expf(mrow[r] - mnew);
      mrow[r] = mnew;
      lrow[r] *= corr;
      ps[r] = 0.f;
#pragma unroll
      for (int t = 0; t < 8; ++t) O[t].f[r] *= corr;
    }
#pragma unroll
    for (int t = 0; t < 4; ++t)
#pragma unroll
      for (int r = 0; r < 8; ++r) {
        float p = __expf(sacc[t].f[r] - mrow[r]);
        sacc[t].f[r] = p;
        ps[r] += p;
      }
#pragma unroll
    for (int r = 0; r < 8; ++r) {
#pragma unroll
      for (int m = 1; m < 16; m <<= 1) ps[r] += __shfl_xor(ps[r], m, 32);
      lrow[r] += ps[r];
    }

#pragma unroll
    for (int t = 0; t < 4; ++t)
#pragma unroll
      for (int r = 0; r < 8; ++r)
        Pst[wave][(r + 8 * half) * VSW_S + t * 16 + nn] = f2bf(sacc[t].f[r]);

#pragma unroll
    for (int kk = 0; kk < 2; ++kk) {
      v16bf a = load_frag_a(Pst[wave] + nn * VSW_S + kk * 32, half);
#pragma unroll
      for (int t = 0; t < 8; ++t) {
        v16bf b = load_frag_b(Vsw + (t * 16 + nn) * VSW_S + kk * 32, half);
        O[t].v = wmma_bf16(a, b, O[t].v);
      }
    }
  }
  unsigned short* crow = ctxb + (size_t)(qrow0 + 8 * half) * D_DIM + h * DHD + nn;
#pragma unroll
  for (int r = 0; r < 8; ++r) {
    float rinv = 1.f / lrow[r];
#pragma unroll
    for (int t = 0; t < 8; ++t) crow[t * 16] = f2bf(O[t].f[r] * rinv);
    crow += D_DIM;
  }
}

// ============================================================
// Router: logits = xb2 @ router_w; softmax; top-2; atomics
// ============================================================
__global__ __launch_bounds__(256) void router_kernel(
    const unsigned short* __restrict__ xb2, const float* __restrict__ rw,
    float* __restrict__ psum, int* __restrict__ counts,
    int* __restrict__ topi, float* __restrict__ topv)
{
  int t = blockIdx.x;
  float lac[NEXP];
#pragma unroll
  for (int e = 0; e < NEXP; ++e) lac[e] = 0.f;
#pragma unroll
  for (int i = 0; i < D_DIM / 256; ++i) {
    int c = threadIdx.x + i * 256;
    float x = bf2f(xb2[(size_t)t * D_DIM + c]);
#pragma unroll
    for (int e = 0; e < NEXP; ++e) lac[e] += x * rw[c * NEXP + e];
  }
  __shared__ float red[8][NEXP];
#pragma unroll
  for (int e = 0; e < NEXP; ++e) {
    float v = lac[e];
#pragma unroll
    for (int off = 16; off > 0; off >>= 1) v += __shfl_down(v, off, 32);
    if ((threadIdx.x & 31) == 0) red[threadIdx.x >> 5][e] = v;
  }
  __syncthreads();
  if (threadIdx.x == 0) {
    float lg[NEXP];
    for (int e = 0; e < NEXP; ++e) {
      float s = 0.f;
      for (int w = 0; w < 8; ++w) s += red[w][e];
      lg[e] = s;
    }
    float mx = lg[0];
    for (int e = 1; e < NEXP; ++e) mx = fmaxf(mx, lg[e]);
    float p[NEXP], se = 0.f;
    for (int e = 0; e < NEXP; ++e) { p[e] = __expf(lg[e] - mx); se += p[e]; }
    float inv = 1.f / se;
    for (int e = 0; e < NEXP; ++e) p[e] *= inv;
    int i0 = 0;
    for (int e = 1; e < NEXP; ++e) if (p[e] > p[i0]) i0 = e;
    int i1 = (i0 == 0) ? 1 : 0;
    for (int e = 0; e < NEXP; ++e) if (e != i0 && p[e] > p[i1]) i1 = e;
    topi[2 * t] = i0;     topv[2 * t] = p[i0];
    topi[2 * t + 1] = i1; topv[2 * t + 1] = p[i1];
    atomicAdd(&counts[i0], 1);
    atomicAdd(&counts[i1], 1);
    for (int e = 0; e < NEXP; ++e) atomicAdd(&psum[e], p[e]);
  }
}

__global__ void finalize_router_kernel(const int* __restrict__ counts,
                                       const float* __restrict__ psum,
                                       const float* __restrict__ aux_in,
                                       int* __restrict__ offsets,
                                       float* __restrict__ out_aux)
{
  if (threadIdx.x == 0 && blockIdx.x == 0) {
    const int T = S_LEN, K = 2;
    int acc = 0;
    float lbl = 0.f;
    for (int e = 0; e < NEXP; ++e) {
      offsets[e] = acc;
      acc += counts[e];
      lbl += ((float)counts[e] / (float)(T * K)) * (psum[e] / (float)T);
    }
    *out_aux = aux_in[0] + (float)NEXP * lbl;
  }
}

__global__ void scatter_kernel(const int* __restrict__ topi, const float* __restrict__ topv,
                               const int* __restrict__ offsets, int* __restrict__ cursor,
                               int* __restrict__ tok_list, float* __restrict__ gate_list,
                               int* __restrict__ slot_of, int T)
{
  int t = blockIdx.x * blockDim.x + threadIdx.x;
  if (t >= T) return;
#pragma unroll
  for (int k = 0; k < 2; ++k) {
    int e = topi[2 * t + k];
    int pos = atomicAdd(&cursor[e], 1);
    int idx = offsets[e] + pos;
    tok_list[idx] = t;
    gate_list[idx] = topv[2 * t + k];
    slot_of[2 * t + k] = idx;
  }
}

// ============================================================
// MoE GEMM1+3 fused: act = silu(xg @ w1[e]) * (xg @ w3[e])
// ============================================================
__global__ __launch_bounds__(256) void moe_gemm13_kernel(
    const unsigned short* __restrict__ xb2, const float* __restrict__ w1,
    const float* __restrict__ w3, const int* __restrict__ tok_list,
    const int* __restrict__ counts, const int* __restrict__ offsets,
    unsigned short* __restrict__ act)
{
  int e = blockIdx.z;
  int cnt = counts[e];
  int r0 = blockIdx.x * TM;
  if (r0 >= cnt) return;
  int off = offsets[e];
  int n0 = blockIdx.y * TN;
  const float* W1 = w1 + (size_t)e * D_DIM * F_DIM;
  const float* W3 = w3 + (size_t)e * D_DIM * F_DIM;

  __shared__ unsigned short As[TM * LDA_S];
  __shared__ unsigned short B1t[TN * LDB_T];
  __shared__ unsigned short B3t[TN * LDB_T];
  const int tid = threadIdx.x;
  const int wave = tid >> 5, lane = tid & 31;
  const int half = lane >> 4, nn = lane & 15;

  const int ar = tid >> 2, aq = tid & 3;
  int gr0 = r0 + ar;      if (gr0 >= cnt) gr0 = cnt - 1;
  int gr1 = r0 + ar + 64; if (gr1 >= cnt) gr1 = cnt - 1;
  const unsigned short* asrc0 = xb2 + (size_t)tok_list[off + gr0] * D_DIM + aq * 8;
  const unsigned short* asrc1 = xb2 + (size_t)tok_list[off + gr1] * D_DIM + aq * 8;
  unsigned short* adst0 = As + ar * LDA_S + aq * 8;
  unsigned short* adst1 = As + (ar + 64) * LDA_S + aq * 8;

  const int nB = tid & 63, r2b = tid >> 6;
  const float* b1src = W1 + (size_t)(2 * r2b) * F_DIM + n0 + nB;
  const float* b3src = W3 + (size_t)(2 * r2b) * F_DIM + n0 + nB;
  unsigned int* b1dst_base = (unsigned int*)(B1t + nB * LDB_T + 2 * r2b);
  unsigned int* b3dst_base = (unsigned int*)(B3t + nB * LDB_T + 2 * r2b);

  CFrag a1[4], a3[4];
#pragma unroll
  for (int t = 0; t < 4; ++t)
#pragma unroll
    for (int j = 0; j < 8; ++j) { a1[t].f[j] = 0.f; a3[t].f[j] = 0.f; }

  for (int k0 = 0; k0 < D_DIM; k0 += TK) {
    stage_b128(adst0, asrc0);
    stage_b128(adst1, asrc1);
    asrc0 += TK; asrc1 += TK;
    unsigned int* b1dst = b1dst_base;
    unsigned int* b3dst = b3dst_base;
#pragma unroll
    for (int it = 0; it < 4; ++it) {
      *b1dst = pack2bf(b1src[0], b1src[F_DIM]);
      *b3dst = pack2bf(b3src[0], b3src[F_DIM]);
      b1src += (size_t)8 * F_DIM; b3src += (size_t)8 * F_DIM;
      b1dst += 4; b3dst += 4;
    }
    if (k0 + TK < D_DIM) {
      __builtin_prefetch(b1src, 0, 1);
      __builtin_prefetch(b3src, 0, 1);
    }
    stage_wait();
    __syncthreads();
    v16bf a = load_frag_a(As + (wave * 16 + nn) * LDA_S, half);
#pragma unroll
    for (int t = 0; t < 4; ++t) {
      v16bf b1 = load_frag_b(B1t + (t * 16 + nn) * LDB_T, half);
      a1[t].v = wmma_bf16(a, b1, a1[t].v);
      v16bf b3 = load_frag_b(B3t + (t * 16 + nn) * LDB_T, half);
      a3[t].v = wmma_bf16(a, b3, a3[t].v);
    }
    __syncthreads();
  }
  const int growbase = r0 + wave * 16 + 8 * half;
  unsigned short* arow = act + (size_t)(off + growbase) * F_DIM + n0 + nn;
#pragma unroll
  for (int r = 0; r < 8; ++r) {
    if (growbase + r < cnt) {
#pragma unroll
      for (int t = 0; t < 4; ++t) {
        float h1 = a1[t].f[r];
        float g = h1 / (1.f + __expf(-h1)) * a3[t].f[r];
        arow[t * 16] = f2bf(g);
      }
    }
    arow += F_DIM;
  }
}

// ============================================================
// MoE GEMM2: y_slot = act @ w2[e]
// ============================================================
__global__ __launch_bounds__(256) void moe_gemm2_kernel(
    const unsigned short* __restrict__ act, const float* __restrict__ w2,
    const int* __restrict__ counts, const int* __restrict__ offsets,
    float* __restrict__ y_slot)
{
  int e = blockIdx.z;
  int cnt = counts[e];
  int r0 = blockIdx.x * TM;
  if (r0 >= cnt) return;
  int off = offsets[e];
  int n0 = blockIdx.y * TN;
  const float* W2 = w2 + (size_t)e * F_DIM * D_DIM;

  __shared__ unsigned short As[TM * LDA_S];
  __shared__ unsigned short Bt[TN * LDB_T];
  const int tid = threadIdx.x;
  const int wave = tid >> 5, lane = tid & 31;
  const int half = lane >> 4, nn = lane & 15;

  const int ar = tid >> 2, aq = tid & 3;
  int gr0 = r0 + ar;      if (gr0 >= cnt) gr0 = cnt - 1;
  int gr1 = r0 + ar + 64; if (gr1 >= cnt) gr1 = cnt - 1;
  const unsigned short* asrc0 = act + (size_t)(off + gr0) * F_DIM + aq * 8;
  const unsigned short* asrc1 = act + (size_t)(off + gr1) * F_DIM + aq * 8;
  unsigned short* adst0 = As + ar * LDA_S + aq * 8;
  unsigned short* adst1 = As + (ar + 64) * LDA_S + aq * 8;

  const int nB = tid & 63, r2b = tid >> 6;
  const float* bsrc = W2 + (size_t)(2 * r2b) * D_DIM + n0 + nB;
  unsigned int* bdst_base = (unsigned int*)(Bt + nB * LDB_T + 2 * r2b);

  CFrag acc[4];
#pragma unroll
  for (int t = 0; t < 4; ++t)
#pragma unroll
    for (int j = 0; j < 8; ++j) acc[t].f[j] = 0.f;

  for (int k0 = 0; k0 < F_DIM; k0 += TK) {
    stage_b128(adst0, asrc0);
    stage_b128(adst1, asrc1);
    asrc0 += TK; asrc1 += TK;
    unsigned int* bdst = bdst_base;
#pragma unroll
    for (int it = 0; it < 4; ++it) {
      *bdst = pack2bf(bsrc[0], bsrc[D_DIM]);
      bsrc += (size_t)8 * D_DIM;
      bdst += 4;
    }
    if (k0 + TK < F_DIM) __builtin_prefetch(bsrc, 0, 1);
    stage_wait();
    __syncthreads();
    v16bf a = load_frag_a(As + (wave * 16 + nn) * LDA_S, half);
#pragma unroll
    for (int t = 0; t < 4; ++t) {
      v16bf b = load_frag_b(Bt + (t * 16 + nn) * LDB_T, half);
      acc[t].v = wmma_bf16(a, b, acc[t].v);
    }
    __syncthreads();
  }
  const int growbase = r0 + wave * 16 + 8 * half;
  float* yrow = y_slot + (size_t)(off + growbase) * D_DIM + n0 + nn;
#pragma unroll
  for (int r = 0; r < 8; ++r) {
    if (growbase + r < cnt) {
#pragma unroll
      for (int t = 0; t < 4; ++t) yrow[t * 16] = acc[t].f[r];
    }
    yrow += D_DIM;
  }
}

// out = h2 + g0*y_slot[s0] + g1*y_slot[s1]
__global__ __launch_bounds__(256) void combine_kernel(
    const float* __restrict__ h2, const float* __restrict__ y_slot,
    const int* __restrict__ slot_of, const float* __restrict__ topv,
    float* __restrict__ out)
{
  int t = blockIdx.x;
  int s0 = slot_of[2 * t], s1 = slot_of[2 * t + 1];
  float g0 = topv[2 * t], g1 = topv[2 * t + 1];
  size_t b = (size_t)t * D_DIM;
#pragma unroll
  for (int i = 0; i < D_DIM / 256; ++i) {
    int c = threadIdx.x + i * 256;
    out[b + c] = h2[b + c] + g0 * y_slot[(size_t)s0 * D_DIM + c]
                           + g1 * y_slot[(size_t)s1 * D_DIM + c];
  }
}

__global__ void mask_fill_kernel(float* __restrict__ m, int n) {
  int i = blockIdx.x * blockDim.x + threadIdx.x;
  if (i < n) m[i] = 1.0f;
}

// ============================================================
// host launcher
// ============================================================
extern "C" void kernel_launch(void* const* d_in, const int* in_sizes, int n_in,
                              void* d_out, int out_size, void* d_ws, size_t ws_size,
                              hipStream_t stream) {
  (void)in_sizes; (void)n_in; (void)out_size; (void)ws_size;
  const float* hidden   = (const float*)d_in[0];
  const float* aux_in   = (const float*)d_in[2];
  const float* ln1      = (const float*)d_in[3];
  const float* qkv_w    = (const float*)d_in[4];
  const float* o_w      = (const float*)d_in[5];
  const float* ln2      = (const float*)d_in[6];
  const float* router_w = (const float*)d_in[7];
  const float* w1       = (const float*)d_in[8];
  const float* w3       = (const float*)d_in[9];
  const float* w2       = (const float*)d_in[10];
  float* out = (float*)d_out;

  char* ws = (char*)d_ws;
  const size_t MB = 1ull << 20;
  float*          qkv   = (float*)(ws + 0);
  float*          oproj = (float*)(ws + 0);
  unsigned short* act   = (unsigned short*)(ws + 16 * MB);
  unsigned short* xb    = (unsigned short*)(ws + 48 * MB);
  unsigned short* qb    = (unsigned short*)(ws + 56 * MB);
  unsigned short* kb    = (unsigned short*)(ws + 64 * MB);
  unsigned short* vb    = (unsigned short*)(ws + 72 * MB);
  unsigned short* ctxb  = (unsigned short*)(ws + 80 * MB);
  float*          y_slot= (float*)(ws + 56 * MB);
  float*          h2    = (float*)(ws + 88 * MB);
  unsigned short* xb2   = (unsigned short*)(ws + 104 * MB);
  char* ctrl = ws + 112 * MB;
  int*   counts  = (int*)(ctrl + 0);
  int*   cursor  = (int*)(ctrl + 32);
  int*   offsets = (int*)(ctrl + 64);
  float* psum    = (float*)(ctrl + 96);
  int*   topi    = (int*)(ctrl + 4096);
  float* topv    = (float*)(ctrl + 4096 + 16384);
  int*   slot_of = (int*)(ctrl + 4096 + 2 * 16384);
  int*   tok_list= (int*)(ctrl + 4096 + 3 * 16384);
  float* gate_list=(float*)(ctrl + 4096 + 4 * 16384);
  (void)gate_list;

  hipMemsetAsync(ctrl, 0, 128, stream);

  rmsnorm_cast_kernel<<<S_LEN, 256, 0, stream>>>(hidden, ln1, xb);
  gemm_bf16_kernel<<<dim3(S_LEN / TM, (3 * D_DIM) / TN), 256, 0, stream>>>(
      xb, qkv_w, qkv, 3 * D_DIM, D_DIM);
  rope_split_kernel<<<dim3(S_LEN, NH), 64, 0, stream>>>(qkv, qb, kb, vb);
  flash_attn_kernel<<<dim3(S_LEN / 64, NH), 128, 0, stream>>>(qb, kb, vb, ctxb);
  gemm_bf16_kernel<<<dim3(S_LEN / TM, D_DIM / TN), 256, 0, stream>>>(
      ctxb, o_w, oproj, D_DIM, D_DIM);
  add_rmsnorm_kernel<<<S_LEN, 256, 0, stream>>>(hidden, oproj, ln2, h2, xb2);

  router_kernel<<<S_LEN, 256, 0, stream>>>(xb2, router_w, psum, counts, topi, topv);
  finalize_router_kernel<<<1, 32, 0, stream>>>(
      counts, psum, aux_in, offsets, out + (size_t)S_LEN * D_DIM + S_LEN);
  scatter_kernel<<<(S_LEN + 255) / 256, 256, 0, stream>>>(
      topi, topv, offsets, cursor, tok_list, gate_list, slot_of, S_LEN);
  moe_gemm13_kernel<<<dim3(S_LEN / TM, F_DIM / TN, NEXP), 256, 0, stream>>>(
      xb2, w1, w3, tok_list, counts, offsets, act);
  moe_gemm2_kernel<<<dim3(S_LEN / TM, D_DIM / TN, NEXP), 256, 0, stream>>>(
      act, w2, counts, offsets, y_slot);
  combine_kernel<<<S_LEN, 256, 0, stream>>>(h2, y_slot, slot_of, topv, out);
  mask_fill_kernel<<<(S_LEN + 255) / 256, 256, 0, stream>>>(
      out + (size_t)S_LEN * D_DIM, S_LEN);
}
